// OptimalTransportCrossAttention_67207648248383
// MI455X (gfx1250) — compile-verified, matrix operand order checked by reading, not run on hardware
//
#include <hip/hip_runtime.h>

#define H_DIM 1024
#define SQ    2048
#define SK    2048
#define NB    16
#define NROWS (NB * SQ)   // 32768

typedef __attribute__((ext_vector_type(16))) __bf16        v16bf;
typedef __attribute__((ext_vector_type(8)))  float         v8f;
typedef __attribute__((ext_vector_type(8)))  unsigned int  v8u;
typedef __attribute__((ext_vector_type(4)))  unsigned int  u32x4;
typedef __attribute__((ext_vector_type(8)))  int           i32x8;
typedef __attribute__((ext_vector_type(4)))  int           i32x4;

#ifndef __has_builtin
#define __has_builtin(x) 0
#endif
#if __has_builtin(__builtin_amdgcn_tensor_load_to_lds) && \
    __has_builtin(__builtin_amdgcn_s_wait_tensorcnt)
#define USE_TDM 1
#else
#define USE_TDM 0
#endif

// ---------------- bf16 helpers (bit-level, RNE) ----------------
__device__ __forceinline__ unsigned short f32_to_bf16(float f) {
  unsigned int u = __builtin_bit_cast(unsigned int, f);
  u += 0x7FFFu + ((u >> 16) & 1u);
  return (unsigned short)(u >> 16);
}
__device__ __forceinline__ unsigned int pack_bf16x2(float lo, float hi) {
  return (unsigned int)f32_to_bf16(lo) | ((unsigned int)f32_to_bf16(hi) << 16);
}
__device__ __forceinline__ float bf16_to_f32(unsigned int h) {
  return __builtin_bit_cast(float, (h & 0xFFFFu) << 16);
}
__device__ __forceinline__ v8f zero8() {
  v8f z = {0.f, 0.f, 0.f, 0.f, 0.f, 0.f, 0.f, 0.f};
  return z;
}

// ---------------- WMMA fragment helpers ----------------
// LDS tiles: rows x 32 bf16 (16 dwords) with row stride 17 dwords
// (17 coprime 64 -> conflict-free frag reads).
// A-frag 16x32 bf16 (ISA 7.12.2): lane m = lane&15 holds row m;
// lanes 0-15 K=0..7,16..23 ; lanes 16-31 K=8..15,24..31 (dword pairs).
// B operand (32x16 K-major) mirrors with N in place of M, so the same
// loader works when the tile is staged N-major.
__device__ __forceinline__ v8u lds_load_frag(const unsigned int* base_dw) {
  const int lane = threadIdx.x & 31;
  const int m    = lane & 15;
  const int c0   = (lane >> 4) << 2;   // dword offset 0 or 4 (elem 0 or 8)
  const unsigned int* row = base_dw + m * 17;
  v8u d;
  d[0] = row[c0 + 0]; d[1] = row[c0 + 1]; d[2] = row[c0 + 2]; d[3] = row[c0 + 3];
  d[4] = row[c0 + 8]; d[5] = row[c0 + 9]; d[6] = row[c0 +10]; d[7] = row[c0 +11];
  return d;
}

__device__ __forceinline__ v8f wmma_bf16(v8u a, v8u b, v8f c) {
  // (neg_a, A, neg_b, B, c_mod, C, reuse_a, reuse_b)
  return __builtin_amdgcn_wmma_f32_16x16x32_bf16(
      false, __builtin_bit_cast(v16bf, a),
      false, __builtin_bit_cast(v16bf, b),
      (short)0, c, false, false);
}

#if USE_TDM
// ---------------- Tensor Data Mover: 2-D bf16 tile -> LDS ----------------
// Loads a (rows x 32)-element bf16 tile (64B rows) into LDS with a 1-dword
// pad after every 16 dwords => 17-dword row stride, matching lds_load_frag.
// D# per CDNA5 ISA.
__device__ __forceinline__ void tdm_load_tile_bf16(
    unsigned int lds_byte_off, const void* gptr, int rows, int stride_elems) {
  const unsigned long long ga = (unsigned long long)(size_t)gptr;
  u32x4 g0;
  g0[0] = 1u;                                   // count=1 (valid), user D#
  g0[1] = lds_byte_off;                         // lds_addr
  g0[2] = (unsigned int)ga;                     // global_addr[31:0]
  g0[3] = (unsigned int)((ga >> 32) & 0x1FFFFFFu) | (2u << 30); // addr[56:32], type=2
  i32x8 g1;
  // data_size=1 (2B), pad_enable=1, pad_interval=3 (16 dwords), pad_amount=0 (1 dword)
  g1[0] = (int)((1u << 16) | (1u << 20) | (3u << 22));
  g1[1] = (int)(((unsigned)stride_elems & 0xFFFFu) << 16);       // tensor_dim0[15:0]
  g1[2] = (int)((((unsigned)stride_elems >> 16) & 0xFFFFu) |
                (((unsigned)rows & 0xFFFFu) << 16));             // dim0 hi | tensor_dim1 lo
  g1[3] = (int)(32u << 16);                                      // tensor_dim1 hi=0 | tile_dim0=32
  g1[4] = rows;                                                  // tile_dim1 | tile_dim2=0
  g1[5] = stride_elems;                                          // tensor_dim0_stride[31:0]
  g1[6] = 0;                                                     // stride hi | dim1_stride lo
  g1[7] = 0;
  i32x4 z4 = {0, 0, 0, 0};
#if __clang_major__ >= 23
  i32x8 z8 = {0, 0, 0, 0, 0, 0, 0, 0};
  __builtin_amdgcn_tensor_load_to_lds(g0, g1, z4, z4, z8, 0);
#else
  __builtin_amdgcn_tensor_load_to_lds(g0, g1, z4, z4, 0);
#endif
}
#endif // USE_TDM

// =====================================================================
// 1) Projection GEMM: Out[r,h] = sum_d X[r,d]*W[d,h] + bias[h]   (fp32 out)
//    Block tile 128(M) x 64(N), K-step 32, bf16 WMMA, fp32 accum.
//    (A tile needs fp32->bf16 conversion, so staged on the VALU.)
// =====================================================================
__global__ __launch_bounds__(256) void proj_gemm(
    const float* __restrict__ X, const float* __restrict__ W,
    const float* __restrict__ bias, float* __restrict__ Out) {
  __shared__ unsigned int sA[128 * 17];
  __shared__ unsigned int sB[64 * 17];
  const int tid  = threadIdx.x;
  const int wave = tid >> 5;
  const int m0   = blockIdx.x * 128;
  const int n0   = blockIdx.y * 64;
  const int wm   = (wave >> 1) * 32;   // 0,32,64,96
  const int wn   = (wave & 1) * 32;    // 0,32

  v8f acc[2][2];
  acc[0][0] = zero8(); acc[0][1] = zero8();
  acc[1][0] = zero8(); acc[1][1] = zero8();

  for (int kt = 0; kt < H_DIM; kt += 32) {
    // stage A: 128x32 fp32 -> bf16 (row-major, stride 17 dw)
    {
      const int r    = tid >> 1;
      const int half = tid & 1;
      const float* src = X + (size_t)(m0 + r) * H_DIM + kt + half * 16;
      unsigned int* dst = sA + r * 17 + half * 8;
#pragma unroll
      for (int i = 0; i < 8; ++i)
        dst[i] = pack_bf16x2(src[2 * i], src[2 * i + 1]);
    }
    // stage B: W[kt..kt+32, n0..n0+64) transposed -> sB[n][k] (N-major)
#pragma unroll
    for (int j = 0; j < 8; ++j) {
      int e = tid + 256 * j;          // 0..2047
      int d = e >> 6;                 // k-row 0..31
      int h = e & 63;                 // n-col 0..63
      float f = W[(size_t)(kt + d) * H_DIM + n0 + h];
      ((unsigned short*)sB)[h * 34 + d] = f32_to_bf16(f);
    }
    __syncthreads();
    v8u a0 = lds_load_frag(sA + (wm) * 17);
    v8u a1 = lds_load_frag(sA + (wm + 16) * 17);
    v8u b0 = lds_load_frag(sB + (wn) * 17);
    v8u b1 = lds_load_frag(sB + (wn + 16) * 17);
    acc[0][0] = wmma_bf16(a0, b0, acc[0][0]);
    acc[0][1] = wmma_bf16(a0, b1, acc[0][1]);
    acc[1][0] = wmma_bf16(a1, b0, acc[1][0]);
    acc[1][1] = wmma_bf16(a1, b1, acc[1][1]);
    __syncthreads();
  }
  const int lane = tid & 31;
  const int cn   = lane & 15;
  const int mb   = (lane >> 4) << 3;
#pragma unroll
  for (int i = 0; i < 2; ++i)
#pragma unroll
    for (int j = 0; j < 2; ++j) {
      const int col = n0 + wn + j * 16 + cn;
      const float bv = bias[col];
      float* dst = Out + (size_t)(m0 + wm + i * 16 + mb) * H_DIM + col;
#pragma unroll
      for (int r = 0; r < 8; ++r)
        dst[(size_t)r * H_DIM] = acc[i][j][r] + bv;
    }
}

// =====================================================================
// 2) Row LayerNorm -> bf16 output + row sum-of-squares (for cdist)
// =====================================================================
__global__ __launch_bounds__(256) void layernorm_rows(
    const float* __restrict__ In, const float* __restrict__ g,
    const float* __restrict__ be, unsigned short* __restrict__ Obf,
    float* __restrict__ rownorm2) {
  __shared__ float red[256];
  const int row = blockIdx.x;
  const int tid = threadIdx.x;
  const float* x = In + (size_t)row * H_DIM;
  float v[4];
  float s = 0.f;
#pragma unroll
  for (int j = 0; j < 4; ++j) { v[j] = x[tid + 256 * j]; s += v[j]; }
  red[tid] = s; __syncthreads();
  for (int o = 128; o > 0; o >>= 1) { if (tid < o) red[tid] += red[tid + o]; __syncthreads(); }
  const float mu = red[0] * (1.0f / (float)H_DIM);
  __syncthreads();
  float s2 = 0.f;
#pragma unroll
  for (int j = 0; j < 4; ++j) { float d = v[j] - mu; s2 += d * d; }
  red[tid] = s2; __syncthreads();
  for (int o = 128; o > 0; o >>= 1) { if (tid < o) red[tid] += red[tid + o]; __syncthreads(); }
  const float rstd = rsqrtf(red[0] * (1.0f / (float)H_DIM) + 1e-5f);
  __syncthreads();
  float q = 0.f;
#pragma unroll
  for (int j = 0; j < 4; ++j) {
    const int c = tid + 256 * j;
    float y = (v[j] - mu) * rstd * g[c] + be[c];
    unsigned short h = f32_to_bf16(y);
    Obf[(size_t)row * H_DIM + c] = h;
    float yr = bf16_to_f32(h);          // norm of the rounded values
    q += yr * yr;
  }
  red[tid] = q; __syncthreads();
  for (int o = 128; o > 0; o >>= 1) { if (tid < o) red[tid] += red[tid + o]; __syncthreads(); }
  if (tid == 0) rownorm2[row] = red[0];
}

// =====================================================================
// 3) Cost/kernel matrix: K[b,q,k] = exp(-4*||qp-kp||) + 1e-12 (bf16)
//    Gram via WMMA. Tile staging is a pure bf16 copy -> Tensor Data
//    Mover DMA with LDS padding (17-dword stride), double-buffered so
//    the DMA of step kt+32 overlaps the WMMAs of step kt.
// =====================================================================
__global__ __launch_bounds__(256) void cost_kernel(
    const unsigned short* __restrict__ Qp, const unsigned short* __restrict__ Kp,
    const float* __restrict__ q2, const float* __restrict__ k2,
    unsigned short* __restrict__ Kmat) {
  __shared__ unsigned int sA[2][128 * 17];
  __shared__ unsigned int sB[2][64 * 17];
  const int tid  = threadIdx.x;
  const int b    = blockIdx.z;
  const int q0   = blockIdx.x * 128;
  const int k0   = blockIdx.y * 64;
  const int wave = tid >> 5;
  const int wm   = (wave >> 1) * 32;
  const int wn   = (wave & 1) * 32;
  const unsigned short* Qbase = Qp + ((size_t)b * SQ + q0) * H_DIM;
  const unsigned short* Kbase = Kp + ((size_t)b * SK + k0) * H_DIM;

#if USE_TDM
  // one wave issues the tile DMAs; TENSORcnt drained before the barrier
  auto stage = [&](int buf, int kt) {
    if (wave == 0) {
      tdm_load_tile_bf16((unsigned int)(size_t)&sA[buf][0], Qbase + kt, 128, H_DIM);
      tdm_load_tile_bf16((unsigned int)(size_t)&sB[buf][0], Kbase + kt, 64, H_DIM);
    }
  };
  auto stage_drain = [&]() {
    if (wave == 0) __builtin_amdgcn_s_wait_tensorcnt(0);
  };
#else
  const unsigned int* Qd = (const unsigned int*)Qbase;
  const unsigned int* Kd = (const unsigned int*)Kbase;
  auto stage = [&](int buf, int kt) {
#pragma unroll
    for (int j = 0; j < 8; ++j) {              // A: 128 x 16 dwords
      int e = tid + 256 * j;
      int r = e >> 4, c = e & 15;
      sA[buf][r * 17 + c] = Qd[(size_t)r * (H_DIM / 2) + (kt >> 1) + c];
    }
#pragma unroll
    for (int j = 0; j < 4; ++j) {              // B: 64 x 16 dwords (N-major already)
      int e = tid + 256 * j;
      int r = e >> 4, c = e & 15;
      sB[buf][r * 17 + c] = Kd[(size_t)r * (H_DIM / 2) + (kt >> 1) + c];
    }
  };
  auto stage_drain = [&]() {};
#endif

  v8f acc[2][2];
  acc[0][0] = zero8(); acc[0][1] = zero8();
  acc[1][0] = zero8(); acc[1][1] = zero8();

  stage(0, 0);
  stage_drain();
  __syncthreads();
  for (int kt = 0; kt < H_DIM; kt += 32) {
    const int cur = (kt >> 5) & 1;
    if (kt + 32 < H_DIM) stage(cur ^ 1, kt + 32);   // overlap DMA with WMMA
    v8u a0 = lds_load_frag(sA[cur] + (wm) * 17);
    v8u a1 = lds_load_frag(sA[cur] + (wm + 16) * 17);
    v8u b0 = lds_load_frag(sB[cur] + (wn) * 17);
    v8u b1 = lds_load_frag(sB[cur] + (wn + 16) * 17);
    acc[0][0] = wmma_bf16(a0, b0, acc[0][0]);
    acc[0][1] = wmma_bf16(a0, b1, acc[0][1]);
    acc[1][0] = wmma_bf16(a1, b0, acc[1][0]);
    acc[1][1] = wmma_bf16(a1, b1, acc[1][1]);
    stage_drain();
    __syncthreads();
  }
  const int lane = tid & 31;
  const int cn   = lane & 15;
  const int mb   = (lane >> 4) << 3;
  const float inv_e2 = 4.0f;   // 1/(EPSILON*EPSILON), EPSILON=0.5
#pragma unroll
  for (int i = 0; i < 2; ++i)
#pragma unroll
    for (int j = 0; j < 2; ++j) {
      const int kk  = k0 + wn + j * 16 + cn;
      const float kv2 = k2[b * SK + kk];
#pragma unroll
      for (int r = 0; r < 8; ++r) {
        const int qq = q0 + wm + i * 16 + mb + r;
        float d2 = q2[b * SQ + qq] + kv2 - 2.0f * acc[i][j][r];
        float cst = sqrtf(fmaxf(d2, 0.0f));
        float kv  = __expf(-cst * inv_e2) + 1e-12f;
        Kmat[((size_t)b * SQ + qq) * SK + kk] = f32_to_bf16(kv);
      }
    }
}

// =====================================================================
// 4) Sinkhorn (bf16 kernel matrix is 134 MB -> resident in 192 MB L2)
// =====================================================================
__global__ void init_uv(float* __restrict__ u, float* __restrict__ v) {
  int i = blockIdx.x * 256 + threadIdx.x;
  u[i] = 1.0f; v[i] = 1.0f;
}

// u[b,q] = a / max(sum_k K[b,q,k]*v[b,k], 1e-8); one block per row.
__global__ __launch_bounds__(256) void sinkhorn_u(
    const unsigned short* __restrict__ Kmat, const float* __restrict__ v,
    float* __restrict__ u) {
  __shared__ float red[256];
  const int row = blockIdx.x;            // 0..NROWS-1
  const int b   = row >> 11;             // row / 2048
  const unsigned int* Kd = (const unsigned int*)(Kmat + (size_t)row * SK);
  const float* vb = v + b * SK;
  const int tid = threadIdx.x;
  float s = 0.f;
#pragma unroll
  for (int j = 0; j < 4; ++j) {
    int idx = tid + 256 * j;             // dword 0..1023
    unsigned int p = Kd[idx];
    s += bf16_to_f32(p) * vb[2 * idx] + bf16_to_f32(p >> 16) * vb[2 * idx + 1];
  }
  red[tid] = s; __syncthreads();
  for (int o = 128; o > 0; o >>= 1) { if (tid < o) red[tid] += red[tid + o]; __syncthreads(); }
  if (tid == 0) {
    const float a = 1.0f / ((float)SQ + 1e-8f);
    u[row] = a / fmaxf(red[0], 1e-8f);
  }
}

// v[b,k] = bb / max(sum_q u[b,q]*K[b,q,k], 1e-8); block = 256 columns.
__global__ __launch_bounds__(256) void sinkhorn_v(
    const unsigned short* __restrict__ Kmat, const float* __restrict__ u,
    float* __restrict__ v) {
  __shared__ float su[SQ];
  const int b  = blockIdx.y;
  const int kk = blockIdx.x * 256 + threadIdx.x;
  const float* ub = u + b * SQ;
#pragma unroll
  for (int j = 0; j < SQ / 256; ++j)
    su[threadIdx.x + 256 * j] = ub[threadIdx.x + 256 * j];
  __syncthreads();
  const unsigned short* Kb = Kmat + (size_t)b * SQ * SK + kk;
  float acc = 0.f;
  for (int q = 0; q < SQ; q += 8) {
    __builtin_prefetch(Kb + (size_t)(q + 8) * SK, 0, 0);
#pragma unroll
    for (int t = 0; t < 8; ++t)
      acc += su[q + t] * bf16_to_f32((unsigned int)Kb[(size_t)(q + t) * SK]);
  }
  const float bb = 1.0f / ((float)SK + 1e-8f);
  v[b * SK + kk] = bb / fmaxf(acc, 1e-8f);
}

// =====================================================================
// 5) Output GEMM: out[b,q,h] = sum_k (u[q]*K[q,k]*v[k]) * value[b,k,h]
// =====================================================================
__global__ __launch_bounds__(256) void ot_output_gemm(
    const unsigned short* __restrict__ Kmat, const float* __restrict__ u,
    const float* __restrict__ v, const float* __restrict__ value,
    float* __restrict__ out) {
  __shared__ unsigned int sA[128 * 17];
  __shared__ unsigned int sB[64 * 17];
  const int tid  = threadIdx.x;
  const int b    = blockIdx.z;
  const int q0   = blockIdx.x * 128;
  const int h0   = blockIdx.y * 64;
  const int wave = tid >> 5;
  const int wm   = (wave >> 1) * 32;
  const int wn   = (wave & 1) * 32;
  const unsigned int* Kd = (const unsigned int*)(Kmat + (size_t)b * SQ * SK);
  const float* ub = u + b * SQ;
  const float* vb = v + b * SK;

  v8f acc[2][2];
  acc[0][0] = zero8(); acc[0][1] = zero8();
  acc[1][0] = zero8(); acc[1][1] = zero8();

  for (int kt = 0; kt < SK; kt += 32) {
    // stage A = (u (x) K (x) v) tile 128x32 -> bf16
    {
      const int r    = tid >> 1;
      const int half = tid & 1;
      const float uq = ub[q0 + r];
      const unsigned int* src = Kd + (size_t)(q0 + r) * (SK / 2) + (kt >> 1) + half * 8;
      unsigned int* dst = sA + r * 17 + half * 8;
#pragma unroll
      for (int i = 0; i < 8; ++i) {
        unsigned int p = src[i];
        const int kc = kt + half * 16 + 2 * i;
        float t0 = bf16_to_f32(p)       * uq * vb[kc];
        float t1 = bf16_to_f32(p >> 16) * uq * vb[kc + 1];
        dst[i] = pack_bf16x2(t0, t1);
      }
    }
    // stage B = value[b, kt..kt+32, h0..h0+64) transposed (N-major, fp32->bf16)
#pragma unroll
    for (int j = 0; j < 8; ++j) {
      int e = tid + 256 * j;
      int d = e >> 6, h = e & 63;
      float f = value[((size_t)b * SK + kt + d) * H_DIM + h0 + h];
      ((unsigned short*)sB)[h * 34 + d] = f32_to_bf16(f);
    }
    __syncthreads();
    v8u a0 = lds_load_frag(sA + (wm) * 17);
    v8u a1 = lds_load_frag(sA + (wm + 16) * 17);
    v8u b0 = lds_load_frag(sB + (wn) * 17);
    v8u b1 = lds_load_frag(sB + (wn + 16) * 17);
    acc[0][0] = wmma_bf16(a0, b0, acc[0][0]);
    acc[0][1] = wmma_bf16(a0, b1, acc[0][1]);
    acc[1][0] = wmma_bf16(a1, b0, acc[1][0]);
    acc[1][1] = wmma_bf16(a1, b1, acc[1][1]);
    __syncthreads();
  }
  const int lane = tid & 31;
  const int cn   = lane & 15;
  const int mb   = (lane >> 4) << 3;
#pragma unroll
  for (int i = 0; i < 2; ++i)
#pragma unroll
    for (int j = 0; j < 2; ++j) {
      const int col = h0 + wn + j * 16 + cn;
      float* dst = out + ((size_t)b * SQ + q0 + wm + i * 16 + mb) * H_DIM + col;
#pragma unroll
      for (int r = 0; r < 8; ++r)
        dst[(size_t)r * H_DIM] = acc[i][j][r];
    }
}

// =====================================================================
// Host-side launch
// =====================================================================
extern "C" void kernel_launch(void* const* d_in, const int* in_sizes, int n_in,
                              void* d_out, int out_size, void* d_ws, size_t ws_size,
                              hipStream_t stream) {
  (void)in_sizes; (void)n_in; (void)out_size; (void)ws_size;
  const float* query = (const float*)d_in[0];
  const float* key   = (const float*)d_in[1];
  const float* value = (const float*)d_in[2];
  const float* qW    = (const float*)d_in[3];
  const float* qb    = (const float*)d_in[4];
  const float* qg    = (const float*)d_in[5];
  const float* qbe   = (const float*)d_in[6];
  const float* kW    = (const float*)d_in[7];
  const float* kb    = (const float*)d_in[8];
  const float* kg    = (const float*)d_in[9];
  const float* kbe   = (const float*)d_in[10];
  float* out = (float*)d_out;

  // workspace carve-up (~400 MB total)
  char* ws = (char*)d_ws;
  size_t off = 0;
  auto carve = [&](size_t bytes) -> void* {
    void* p = ws + off;
    off += (bytes + 255) & ~(size_t)255;
    return p;
  };
  unsigned short* qp   = (unsigned short*)carve((size_t)NROWS * H_DIM * 2); // 64 MB
  unsigned short* kp   = (unsigned short*)carve((size_t)NROWS * H_DIM * 2); // 64 MB
  unsigned short* Kmat = (unsigned short*)carve((size_t)NB * SQ * SK * 2);  // 134 MB (L2-resident)
  float* proj_tmp = (float*)carve((size_t)NROWS * H_DIM * 4);               // 128 MB (reused Q then K)
  float* q2 = (float*)carve((size_t)NROWS * 4);
  float* k2 = (float*)carve((size_t)NROWS * 4);
  float* u  = (float*)carve((size_t)NROWS * 4);
  float* v  = (float*)carve((size_t)NROWS * 4);

  const dim3 blk(256);

  // Q path
  proj_gemm<<<dim3(NROWS / 128, H_DIM / 64), blk, 0, stream>>>(query, qW, qb, proj_tmp);
  layernorm_rows<<<NROWS, blk, 0, stream>>>(proj_tmp, qg, qbe, qp, q2);
  // K path (reuses proj_tmp)
  proj_gemm<<<dim3(NROWS / 128, H_DIM / 64), blk, 0, stream>>>(key, kW, kb, proj_tmp);
  layernorm_rows<<<NROWS, blk, 0, stream>>>(proj_tmp, kg, kbe, kp, k2);

  // cost -> Sinkhorn kernel matrix (bf16), TDM-staged tiles
  cost_kernel<<<dim3(SQ / 128, SK / 64, NB), blk, 0, stream>>>(qp, kp, q2, k2, Kmat);

  // Sinkhorn iterations
  init_uv<<<NROWS / 256, blk, 0, stream>>>(u, v);
  for (int it = 0; it < 50; ++it) {
    sinkhorn_u<<<NROWS, blk, 0, stream>>>(Kmat, v, u);
    sinkhorn_v<<<dim3(SK / 256, NB), blk, 0, stream>>>(Kmat, u, v);
  }

  // out = (u * K * v) @ value
  ot_output_gemm<<<dim3(SQ / 128, H_DIM / 64, NB), blk, 0, stream>>>(Kmat, u, v, value, out);
}
